// ScannedLSTM_68899865362872
// MI455X (gfx1250) — matrix operand order repaired and checked by reading, not compiled, not run on hardware
//
#include <hip/hip_runtime.h>
#include <hip/hip_bf16.h>

// ---------------------------------------------------------------------------
// Scanned LSTM for MI455X (gfx1250, wave32, WMMA).
//   Phase A: convert X/Wi/Wh/h0 to bf16 (weights transposed to [N][K]).
//   Phase B: XWi = X @ Wi  — big parallel bf16 WMMA GEMM, f32 out (ws).
//   Phase C: persistent 32-WG scan kernel: Wh slice staged into LDS via the
//            Tensor Data Mover (TDM), c-state in VGPRs, h broadcast via a
//            double-buffered bf16 global buffer, one fence+atomic grid
//            barrier per timestep.
// ---------------------------------------------------------------------------

typedef __attribute__((ext_vector_type(16))) __bf16 v16bf;
typedef __attribute__((ext_vector_type(8)))  __bf16 v8bf;
typedef __attribute__((ext_vector_type(8)))  float  v8f;
typedef __attribute__((ext_vector_type(4)))  unsigned int u32x4;
typedef __attribute__((ext_vector_type(8)))  int          i32x8;
typedef __attribute__((ext_vector_type(4)))  int          i32x4;

#define LSTM_B   64
#define LSTM_T   1024
#define LSTM_D   512
#define LSTM_H   512
#define LSTM_G4  2048
#define SCAN_NWG 32

__device__ __forceinline__ float fsig(float x)  { return 1.0f / (1.0f + __expf(-x)); }
__device__ __forceinline__ float ftanh(float x) { float e = __expf(-2.0f * x); return (1.0f - e) / (1.0f + e); }

// A fragment (16x32 bf16, row-major source): lane<16 -> row=lane, K = {0..15};
// lane>=16 -> row=lane-16, K = {16..31}.  Two contiguous 16B loads.
__device__ __forceinline__ v16bf load_a_frag(const __bf16* __restrict__ base, int ld,
                                             int m0, int k0, int lane) {
  const __bf16* p = base + (size_t)(m0 + (lane & 15)) * ld + k0 + ((lane >> 4) << 4);
  v8bf lo = *(const v8bf*)p;
  v8bf hi = *(const v8bf*)(p + 8);
  v16bf r;
#pragma unroll
  for (int i = 0; i < 8; ++i) { r[i] = lo[i]; r[i + 8] = hi[i]; }
  return r;
}

__device__ __forceinline__ v8f wmma_bf16(v16bf a, v16bf b, v8f c) {
  return __builtin_amdgcn_wmma_f32_16x16x32_bf16(false, a, false, b, (short)0, c, false, false);
}

// ---------------------------------------------------------------------------
// Conversion kernels
// ---------------------------------------------------------------------------
__global__ void k_f32_to_bf16(const float* __restrict__ s, __bf16* __restrict__ d, int n) {
  for (int i = blockIdx.x * blockDim.x + threadIdx.x; i < n; i += gridDim.x * blockDim.x)
    d[i] = (__bf16)s[i];
}

// s: [rows][cols] f32 (K-major weights)  ->  d: [cols][rows] bf16 ([N][K])
__global__ void k_transpose_to_bf16(const float* __restrict__ s, __bf16* __restrict__ d,
                                    int rows, int cols) {
  int total = rows * cols;
  for (int i = blockIdx.x * blockDim.x + threadIdx.x; i < total; i += gridDim.x * blockDim.x) {
    int n = i / rows;
    int k = i - n * rows;
    d[i] = (__bf16)s[(size_t)k * cols + n];
  }
}

// ---------------------------------------------------------------------------
// Phase B: XWi[M][2048] = Xbf[M][512] @ Wi (WiT given as [2048][512] bf16)
// 128x128 block tile, 256 threads (8 waves), each wave: 16 rows x 8 N-tiles.
// ---------------------------------------------------------------------------
__global__ __launch_bounds__(256) void k_gemm_xwi(const __bf16* __restrict__ Xbf,
                                                  const __bf16* __restrict__ WiT,
                                                  float* __restrict__ XWi) {
  constexpr int K = LSTM_D, N = LSTM_G4;
  __shared__ __bf16 bLds[128][32];  // B panel: 128 cols x 32 K (8 KB)

  const int tid  = threadIdx.x;
  const int lane = tid & 31;
  const int wave = tid >> 5;
  const int n0   = blockIdx.x * 128;
  const int m0   = blockIdx.y * 128;

  v8f acc[8];
#pragma unroll
  for (int i = 0; i < 8; ++i) acc[i] = (v8f){0.f, 0.f, 0.f, 0.f, 0.f, 0.f, 0.f, 0.f};

  for (int k0 = 0; k0 < K; k0 += 32) {
    __syncthreads();
    {   // stage B panel: each thread copies 32 contiguous bytes
      int row  = tid >> 1;
      int half = tid & 1;
      *(v16bf*)&bLds[row][half * 16] =
          *(const v16bf*)(WiT + (size_t)(n0 + row) * K + k0 + half * 16);
    }
    __syncthreads();

    v16bf a = load_a_frag(Xbf, K, m0 + wave * 16, k0, lane);
#pragma unroll
    for (int nt = 0; nt < 8; ++nt) {
      // B fragment: lane holds col = nt*16 + lane%16, 16 contiguous K values
      v16bf b = *(const v16bf*)&bLds[nt * 16 + (lane & 15)][(lane >> 4) << 4];
      acc[nt] = wmma_bf16(a, b, acc[nt]);
    }
  }

#pragma unroll
  for (int nt = 0; nt < 8; ++nt) {
    int col = n0 + nt * 16 + (lane & 15);
#pragma unroll
    for (int v = 0; v < 8; ++v) {
      int row = m0 + wave * 16 + v + ((lane >> 4) << 3);
      XWi[(size_t)row * N + col] = acc[nt][v];
    }
  }
}

// ---------------------------------------------------------------------------
// Grid barrier (fence + L2 atomics); s_cluster_barrier is a NOP when the
// dispatch is not clustered, included to use the CDNA5 HW-barrier path.
// ---------------------------------------------------------------------------
__device__ __forceinline__ void grid_barrier(int* cnt, int* gen, int nwg) {
  __builtin_amdgcn_fence(__ATOMIC_RELEASE, "agent");
  __syncthreads();
  if (threadIdx.x == 0) {
    __builtin_amdgcn_s_cluster_barrier();
    int g = __hip_atomic_load(gen, __ATOMIC_RELAXED, __HIP_MEMORY_SCOPE_AGENT);
    if (__hip_atomic_fetch_add(cnt, 1, __ATOMIC_ACQ_REL, __HIP_MEMORY_SCOPE_AGENT) == nwg - 1) {
      __hip_atomic_store(cnt, 0, __ATOMIC_RELAXED, __HIP_MEMORY_SCOPE_AGENT);
      __hip_atomic_fetch_add(gen, 1, __ATOMIC_RELEASE, __HIP_MEMORY_SCOPE_AGENT);
    } else {
      while (__hip_atomic_load(gen, __ATOMIC_ACQUIRE, __HIP_MEMORY_SCOPE_AGENT) == g)
        __builtin_amdgcn_s_sleep(2);
    }
  }
  __syncthreads();
  __builtin_amdgcn_fence(__ATOMIC_ACQUIRE, "agent");
}

// ---------------------------------------------------------------------------
// Phase C: persistent scan. 32 WGs x 128 threads (4 waves).
//   WG b owns hidden cols [16b, 16b+16); its 4x16 = 64 gate columns of WhT
//   (64 KB bf16) are DMA'd into LDS by the Tensor Data Mover as one 3D tile
//   (K=512 x 16 rows x 4 gates, gate pitch 512*512 elements) and stay
//   resident for all 1024 steps. Wave w owns batch rows [16w, 16w+16) and
//   keeps its 16x16 c-state tile in registers.
// ---------------------------------------------------------------------------
__global__ __launch_bounds__(128) void k_lstm_scan(const __bf16* __restrict__ WhT,  // [2048][512]
                                                   const float* __restrict__ XWi,   // [B*T][2048]
                                                   const float* __restrict__ bias,  // [2048]
                                                   const float* __restrict__ c0,    // [64][512]
                                                   __bf16* __restrict__ hbuf,       // [2][64][512]
                                                   float* __restrict__ outC,
                                                   float* __restrict__ outH,
                                                   float* __restrict__ outY,
                                                   int* bar_cnt, int* bar_gen) {
  constexpr int T = LSTM_T, H = LSTM_H, G4 = LSTM_G4;
  __shared__ __bf16 wh[64][512];  // 64 KB of the 320 KB LDS

  const int tid   = threadIdx.x;
  const int lane  = tid & 31;
  const int wave  = tid >> 5;
  const int hBase = blockIdx.x * 16;

  // ---- TDM: DMA the Wh slice into LDS (wave 0 issues, TENSORcnt tracks) ----
  if (tid < 32) {
    unsigned long long gaddr = (unsigned long long)(uintptr_t)(WhT + (size_t)hBase * H);
    unsigned int       laddr = (unsigned int)(uintptr_t)&wh[0][0];  // LDS byte offset

    u32x4 g0;
    g0[0] = 1u;                                        // count=1 valid descriptor
    g0[1] = laddr;                                     // lds_addr
    g0[2] = (unsigned int)(gaddr & 0xffffffffu);       // global_addr[31:0]
    g0[3] = (unsigned int)(gaddr >> 32) | 0x80000000u; // global_addr[56:32] | type=2

    i32x8 g1;
    g1[0] = 1 << 16;                 // workgroup_mask=0, data_size=1 (2 bytes)
    g1[1] = (H & 0xffff) << 16;      // tensor_dim0[15:0] = 512 (K)
    g1[2] = (H & 0xffff) << 16;      // tensor_dim0 hi=0 | tensor_dim1[15:0] = 512 rows/gate
    g1[3] = (H & 0xffff) << 16;      // tensor_dim1 hi=0 | tile_dim0 = 512
    g1[4] = 16 | (4 << 16);          // tile_dim1 = 16 rows, tile_dim2 = 4 gates
    g1[5] = H;                       // tensor_dim0_stride = 512 (row pitch)
    g1[6] = 0;                       // stride0 hi | tensor_dim1_stride[15:0] (512*512 -> low16=0)
    g1[7] = (H * H) >> 16;           // tensor_dim1_stride[47:16] = 4 (gate pitch 262144)

    i32x4 g2;
    g2[0] = 4;                       // tensor_dim2 = 4 (gates)
    g2[1] = 0;                       // tensor_dim3
    g2[2] = 0;                       // tensor_dim2_stride lo (4th dim unused)
    g2[3] = 0;                       // stride hi | tile_dim3 = 0
    i32x4 g3 = {0, 0, 0, 0};         // dim3 stride / dim4 / tile_dim4
    i32x8 g4 = {0, 0, 0, 0, 0, 0, 0, 0};  // extra group (clang-23 6-arg form), zero-filled

    __builtin_amdgcn_tensor_load_to_lds(g0, g1, g2, g3, g4, 0);
    __builtin_amdgcn_s_wait_tensorcnt(0);
  }

  // c-state tile in registers (C/D layout: row = v + 8*(lane>=16), col = lane%16)
  const int mrowBase = wave * 16 + ((lane >> 4) << 3);
  const int col      = hBase + (lane & 15);
  float c[8];
#pragma unroll
  for (int v = 0; v < 8; ++v) c[v] = c0[(size_t)(mrowBase + v) * H + col];

  float bgate[4];
#pragma unroll
  for (int q = 0; q < 4; ++q) bgate[q] = bias[q * H + col];

  __syncthreads();   // Wh slice now visible in LDS to all waves

  for (int t = 0; t < T; ++t) {
    const __bf16* hc = hbuf + (size_t)(t & 1) * LSTM_B * H;
    __bf16*       hn = hbuf + (size_t)((t + 1) & 1) * LSTM_B * H;

    // acc = XWi[:, t, :] + bias  (precomputed input contribution)
    v8f acc[4];
#pragma unroll
    for (int q = 0; q < 4; ++q) {
      int n = q * H + col;
#pragma unroll
      for (int v = 0; v < 8; ++v)
        acc[q][v] = XWi[((size_t)(mrowBase + v) * T + t) * G4 + n] + bgate[q];
    }
    if (t + 1 < T)
      __builtin_prefetch(&XWi[((size_t)mrowBase * T + t + 1) * G4 + col], 0, 1);

    // acc += h @ Wh[:, gate cols]
#pragma unroll 4
    for (int k0 = 0; k0 < H; k0 += 32) {
      v16bf a = load_a_frag(hc, H, wave * 16, k0, lane);
#pragma unroll
      for (int q = 0; q < 4; ++q) {
        v16bf b = *(const v16bf*)&wh[q * 16 + (lane & 15)][k0 + ((lane >> 4) << 4)];
        acc[q] = wmma_bf16(a, b, acc[q]);
      }
    }

    // pointwise LSTM cell (thread-local: all 4 gates share the lane/VGPR slot)
#pragma unroll
    for (int v = 0; v < 8; ++v) {
      float i_ = fsig(acc[0][v]);
      float f_ = fsig(acc[1][v]);
      float g_ = ftanh(acc[2][v]);
      float o_ = fsig(acc[3][v]);
      float nc = f_ * c[v] + i_ * g_;
      float nh = o_ * ftanh(nc);
      c[v] = nc;
      int brow = mrowBase + v;
      hn[(size_t)brow * H + col]               = (__bf16)nh;  // broadcast for t+1
      outY[((size_t)brow * T + t) * H + col]   = nh;          // ys[b, t, h]
      if (t == T - 1) {
        outC[(size_t)brow * H + col] = nc;
        outH[(size_t)brow * H + col] = nh;
      }
    }

    grid_barrier(bar_cnt, bar_gen, SCAN_NWG);
  }
}

// ---------------------------------------------------------------------------
extern "C" void kernel_launch(void* const* d_in, const int* in_sizes, int n_in,
                              void* d_out, int out_size, void* d_ws, size_t ws_size,
                              hipStream_t stream) {
  (void)in_sizes; (void)n_in; (void)out_size; (void)ws_size;
  const float* x  = (const float*)d_in[0];
  const float* c0 = (const float*)d_in[1];
  const float* h0 = (const float*)d_in[2];
  const float* Wi = (const float*)d_in[3];
  const float* Wh = (const float*)d_in[4];
  const float* b  = (const float*)d_in[5];

  constexpr size_t M = (size_t)LSTM_B * LSTM_T;  // 65536 rows

  char*  ws  = (char*)d_ws;
  size_t off = 0;
  auto alloc = [&](size_t bytes) { size_t o = off; off = (off + bytes + 255) & ~(size_t)255; return o; };

  int*    bar  = (int*)(ws + alloc(256));
  __bf16* Xbf  = (__bf16*)(ws + alloc(M * LSTM_D * 2));
  __bf16* WiT  = (__bf16*)(ws + alloc((size_t)LSTM_G4 * LSTM_D * 2));
  __bf16* WhT  = (__bf16*)(ws + alloc((size_t)LSTM_G4 * LSTM_H * 2));
  __bf16* hbuf = (__bf16*)(ws + alloc((size_t)2 * LSTM_B * LSTM_H * 2));
  float*  XWi  = (float*)(ws + alloc(M * LSTM_G4 * 4));

  float* outC = (float*)d_out;
  float* outH = outC + (size_t)LSTM_B * LSTM_H;
  float* outY = outH + (size_t)LSTM_B * LSTM_H;

  (void)hipMemsetAsync(bar, 0, 256, stream);

  k_f32_to_bf16<<<4096, 256, 0, stream>>>(x, Xbf, (int)(M * LSTM_D));
  k_transpose_to_bf16<<<2048, 256, 0, stream>>>(Wi, WiT, LSTM_D, LSTM_G4);
  k_transpose_to_bf16<<<2048, 256, 0, stream>>>(Wh, WhT, LSTM_H, LSTM_G4);
  k_f32_to_bf16<<<128, 256, 0, stream>>>(h0, hbuf, LSTM_B * LSTM_H);

  dim3 g1(LSTM_G4 / 128, (unsigned)(M / 128));
  k_gemm_xwi<<<g1, 256, 0, stream>>>(Xbf, WiT, XWi);

  k_lstm_scan<<<SCAN_NWG, 128, 0, stream>>>(WhT, XWi, b, c0, hbuf,
                                            outC, outH, outY, bar + 0, bar + 16);
}
